// AutoregressiveSplineDeep_27041114096365
// MI455X (gfx1250) — matrix operand order repaired and checked by reading, not compiled, not run on hardware
//
#include <hip/hip_runtime.h>
#include <hip/hip_bf16.h>

// ---------------------------------------------------------------------------
// AutoregressiveSplineDeep for MI455X (gfx1250, wave32, WMMA)
//
// Fusion strategy: each block owns SPB=64 samples for ALL 8 flow steps.
// Hypernet GEMMs (128x128, 128x126) run on v_wmma_f32_16x16x32_f16 with f32
// accumulation; MADE masks are folded into the f16 weight staging. The
// linear-rational spline runs per (sample,dim) on VALU/transcendentals.
// ---------------------------------------------------------------------------

typedef __attribute__((ext_vector_type(16))) _Float16 v16h;
typedef __attribute__((ext_vector_type(8)))  _Float16 v8h;
typedef __attribute__((ext_vector_type(8)))  float    v8f;

#define HH        128      // hidden width
#define MD        126      // M*D = (4*16-1)*2
#define SPB       64       // samples per block
#define NTHR      256      // 8 waves of 32
#define FLOW_LEN  8
#define SBOUND    5.0f

__device__ __forceinline__ float softplusf(float t) {
    // log1p(exp(t)), numerically stable
    return fmaxf(t, 0.0f) + __logf(1.0f + __expf(-fabsf(t)));
}
__device__ __forceinline__ float sigmoidf(float t) {
    return 1.0f / (1.0f + __expf(-t));
}

// A fragment (16x32 f16, M=samples) from a row-major LDS tile, ld = HH halves.
// Lane l<16 holds row M=l, K = k0+{0..15}; lane l+16 holds K = k0+{16..31}.
__device__ __forceinline__ v16h frag_a(const _Float16* base, int lane, int k0) {
    const int r  = lane & 15;
    const int h8 = (lane >> 4) << 4;          // 0 or 16
    const _Float16* p = base + r * HH + k0 + h8;
    v8h lo = *(const v8h*)(p);                // ds_load_b128
    v8h hi = *(const v8h*)(p + 8);
    v16h a;
#pragma unroll
    for (int i = 0; i < 8; ++i) { a[i] = lo[i]; a[i + 8] = hi[i]; }
    return a;
}

// B fragment (32x16 f16, K x N) from row-major weight LDS W[out][in] (ld=HH).
// Lane (n = lane&15) with ksel = lane>>4 holds K = k0 + ksel*16 + {0..15},
// which is one contiguous 32-byte run of row (n0+n) of W.
__device__ __forceinline__ v16h frag_b(const _Float16* W, int n0, int k0, int lane) {
    const int col = lane & 15;
    const int kh  = (lane >> 4) << 4;         // 0 or 16
    const _Float16* p = W + (n0 + col) * HH + k0 + kh;
    v8h lo = *(const v8h*)(p);
    v8h hi = *(const v8h*)(p + 8);
    v16h b;
#pragma unroll
    for (int i = 0; i < 8; ++i) { b[i] = lo[i]; b[i + 8] = hi[i]; }
    return b;
}

__device__ __forceinline__ v8f wmma16(v16h a, v16h b, v8f c) {
    // v_wmma_f32_16x16x32_f16 (8-arg form: neg_a, A, neg_b, B, c_mod, C, reuse_a, reuse_b)
    return __builtin_amdgcn_wmma_f32_16x16x32_f16(false, a, false, b, (short)0, c, false, false);
}

__global__ __launch_bounds__(NTHR)
void AutoregressiveSplineDeep_kernel(const float* __restrict__ z,
                                     const float* __restrict__ W0,
                                     const float* __restrict__ b0,
                                     const float* __restrict__ W1,
                                     const float* __restrict__ b1,
                                     const float* __restrict__ W2,
                                     const float* __restrict__ b2,
                                     float* __restrict__ out, int n) {
    __shared__ _Float16 sW1[HH * HH];     // masked layer-1 weights (f16)
    __shared__ _Float16 sW2[HH * HH];     // masked layer-2 weights, padded 126->128
    __shared__ _Float16 sActA[SPB * HH];  // h1 activations
    __shared__ _Float16 sActB[SPB * HH];  // h2 activations
    __shared__ float    sOut[SPB * HH];   // layer-2 f32 outputs (spline params)
    __shared__ float    sX[SPB * 2];      // current flow state
    __shared__ float    sB0[HH], sB1[HH], sB2[HH];
    __shared__ float    sW0[HH * 2];      // masked layer-0 weights

    const int tid  = threadIdx.x;
    const int lane = tid & 31;
    const int wv   = tid >> 5;     // wave 0..7
    const int st   = wv & 3;       // sample tile 0..3 (16 samples each)
    const int fh   = wv >> 2;      // feature-tile half 0..1
    const int s0   = blockIdx.x * SPB;

    // Warm the (L2-resident) weights: gfx1250 global_prefetch_b8 path.
    __builtin_prefetch((const char*)W1 + tid * 256, 0, 1);
    __builtin_prefetch((const char*)W2 + tid * 248, 0, 1);

    // ---- Stage masked weights into LDS as f16 (masks folded in) ----
    for (int i = tid; i < HH * HH; i += NTHR) {
        const int r = i >> 7;
        sW1[i] = (_Float16)W1[i];                 // MASK1 == all ones (D=2)
        float w2 = 0.0f;
        if ((r & 1) && (r < MD)) w2 = W2[i];      // MASK2: out_deg>1 <=> odd row
        sW2[i] = (_Float16)w2;                    // rows 126,127 padded to zero
    }
    for (int i = tid; i < HH; i += NTHR) {
        sB0[i] = b0[i];
        sB1[i] = b1[i];
        float v2 = 0.0f;
        if (i < MD) v2 = b2[i];
        sB2[i] = v2;
        sW0[i * 2 + 0] = W0[i * 2 + 0];           // MASK0 col 0 = 1
        sW0[i * 2 + 1] = 0.0f;                    // MASK0 col 1 = 0
    }
    for (int i = tid; i < SPB * 2; i += NTHR) {
        const int g = s0 * 2 + i;
        float xv = 0.0f;
        if (g < n * 2) xv = z[g];
        sX[i] = xv;
    }
    __syncthreads();

#pragma unroll 1
    for (int step = 0; step < FLOW_LEN; ++step) {
        // ---- layer 0 (K=2, VALU): h1 = relu(x @ W0m^T + b0) ----
        for (int i = tid; i < SPB * HH; i += NTHR) {
            const int s = i >> 7, j = i & 127;
            float v = sB0[j] + sW0[j * 2 + 0] * sX[s * 2 + 0]
                            + sW0[j * 2 + 1] * sX[s * 2 + 1];
            sActA[i] = (_Float16)fmaxf(v, 0.0f);
        }
        __syncthreads();

        // ---- layer 1: (64x128)x(128x128)^T via WMMA, relu + bias ----
        {
            const _Float16* abase = sActA + st * 16 * HH;
            v16h a0 = frag_a(abase, lane, 0);
            v16h a1 = frag_a(abase, lane, 32);
            v16h a2 = frag_a(abase, lane, 64);
            v16h a3 = frag_a(abase, lane, 96);
            const int col   = lane & 15;
            const int rbase = st * 16 + ((lane >> 4) << 3);
#pragma unroll
            for (int t = 0; t < 4; ++t) {
                const int nt = fh * 4 + t;
                v8f acc = {};
                acc = wmma16(a0, frag_b(sW1, nt * 16, 0,  lane), acc);
                acc = wmma16(a1, frag_b(sW1, nt * 16, 32, lane), acc);
                acc = wmma16(a2, frag_b(sW1, nt * 16, 64, lane), acc);
                acc = wmma16(a3, frag_b(sW1, nt * 16, 96, lane), acc);
                const int jo = nt * 16 + col;
                const float bias = sB1[jo];
#pragma unroll
                for (int v = 0; v < 8; ++v)
                    sActB[(rbase + v) * HH + jo] = (_Float16)fmaxf(acc[v] + bias, 0.0f);
            }
        }
        __syncthreads();

        // ---- layer 2: (64x128)x(128x128)^T via WMMA, +b2, f32 out ----
        {
            const _Float16* abase = sActB + st * 16 * HH;
            v16h a0 = frag_a(abase, lane, 0);
            v16h a1 = frag_a(abase, lane, 32);
            v16h a2 = frag_a(abase, lane, 64);
            v16h a3 = frag_a(abase, lane, 96);
            const int col   = lane & 15;
            const int rbase = st * 16 + ((lane >> 4) << 3);
#pragma unroll
            for (int t = 0; t < 4; ++t) {
                const int nt = fh * 4 + t;
                v8f acc = {};
                acc = wmma16(a0, frag_b(sW2, nt * 16, 0,  lane), acc);
                acc = wmma16(a1, frag_b(sW2, nt * 16, 32, lane), acc);
                acc = wmma16(a2, frag_b(sW2, nt * 16, 64, lane), acc);
                acc = wmma16(a3, frag_b(sW2, nt * 16, 96, lane), acc);
                const int jo = nt * 16 + col;
                const float bias = sB2[jo];
#pragma unroll
                for (int v = 0; v < 8; ++v)
                    sOut[(rbase + v) * HH + jo] = acc[v] + bias;
            }
        }
        __syncthreads();

        // ---- spline: one thread per (sample, dim) ----
        float xnew = 0.0f;
        if (tid < SPB * 2) {
            const int s = tid >> 1, d = tid & 1;
            const float* P = sOut + s * HH + d;   // param mi lives at P[mi*2]
            const float mb = 1e-3f, mdv = 1e-3f;
            const float c1 = 1.0f - mb * 16.0f;
            float x = sX[tid];

            // softmax(widths), softmax(heights)
            float mxw = P[0], mxh = P[32];
#pragma unroll
            for (int m = 1; m < 16; ++m) {
                mxw = fmaxf(mxw, P[m * 2]);
                mxh = fmaxf(mxh, P[(16 + m) * 2]);
            }
            float ew[16], eh[16], sw = 0.0f, sh = 0.0f;
#pragma unroll
            for (int m = 0; m < 16; ++m) {
                ew[m] = __expf(P[m * 2] - mxw);        sw += ew[m];
                eh[m] = __expf(P[(16 + m) * 2] - mxh); sh += eh[m];
            }
            const float rw = 1.0f / sw, rh = 1.0f / sh;

            // knot positions via cumsum (registers only; constant indices)
            float cw[17], ch[17];
            cw[0] = -SBOUND; ch[0] = -SBOUND;
            float aw = 0.0f, ah = 0.0f;
#pragma unroll
            for (int m = 0; m < 16; ++m) {
                aw += mb + c1 * ew[m] * rw;  cw[m + 1] = aw * (2.0f * SBOUND) - SBOUND;
                ah += mb + c1 * eh[m] * rh;  ch[m + 1] = ah * (2.0f * SBOUND) - SBOUND;
            }
            cw[16] = SBOUND; ch[16] = SBOUND;

            const bool inside = (x >= -SBOUND) && (x <= SBOUND);
            const float xc = fminf(fmaxf(x, -SBOUND), SBOUND);
            int idx = -1;
#pragma unroll
            for (int m = 0; m < 17; ++m) idx += (xc >= cw[m]) ? 1 : 0;
            idx = idx < 0 ? 0 : (idx > 15 ? 15 : idx);

            // select knot values with unrolled cndmask chain (no scratch)
            float xk = 0.0f, xk1 = 0.0f, yk = 0.0f, yk1 = 0.0f;
#pragma unroll
            for (int m = 0; m < 16; ++m) {
                const bool sel = (m == idx);
                xk  = sel ? cw[m]     : xk;
                xk1 = sel ? cw[m + 1] : xk1;
                yk  = sel ? ch[m]     : yk;
                yk1 = sel ? ch[m + 1] : yk1;
            }
            const float wk = xk1 - xk, hk = yk1 - yk;
            const float sk = hk / wk;

            // derivatives / lambda: dynamic-index LDS reads (plain ds_load)
            const float dk  = (idx == 0)  ? 1.0f : mdv + softplusf(P[(32 + idx - 1) * 2]);
            const float dk1 = (idx == 15) ? 1.0f : mdv + softplusf(P[(32 + idx) * 2]);
            const float lk  = 0.95f * sigmoidf(P[(47 + idx) * 2]) + 0.025f;

            const float wb = sqrtf(dk / dk1);
            const float wc = (lk * dk + (1.0f - lk) * wb * dk1) / sk;
            const float ya = yk, yb = yk1;
            const float yc = ((1.0f - lk) * ya + lk * wb * yb)
                           / ((1.0f - lk) + lk * wb);
            const float th = (xc - xk) / wk;
            const bool left = th <= lk;
            const float num = left ? (ya * (lk - th) + wc * yc * th)
                                   : (wc * yc * (1.0f - th) + wb * yb * (th - lk));
            const float den = left ? ((lk - th) + wc * th)
                                   : (wc * (1.0f - th) + wb * (th - lk));
            xnew = inside ? (num / den) : x;
        }
        __syncthreads();
        if (tid < SPB * 2) sX[tid] = xnew;
        __syncthreads();
    }

    // ---- write final state ----
    for (int i = tid; i < SPB * 2; i += NTHR) {
        const int g = s0 * 2 + i;
        if (g < n * 2) out[g] = sX[i];
    }
}

extern "C" void kernel_launch(void* const* d_in, const int* in_sizes, int n_in,
                              void* d_out, int out_size, void* d_ws, size_t ws_size,
                              hipStream_t stream) {
    const float* z  = (const float*)d_in[0];
    const float* W0 = (const float*)d_in[1];
    const float* b0 = (const float*)d_in[2];
    const float* W1 = (const float*)d_in[3];
    const float* b1 = (const float*)d_in[4];
    const float* W2 = (const float*)d_in[5];
    const float* b2 = (const float*)d_in[6];
    float* out = (float*)d_out;

    const int n = in_sizes[0] / 2;              // N samples (D=2)
    const int grid = (n + SPB - 1) / SPB;       // 8192 blocks for N=524288
    AutoregressiveSplineDeep_kernel<<<grid, NTHR, 0, stream>>>(
        z, W0, b0, W1, b1, W2, b2, out, n);
}
